// PointNetSetAbstraction_80771154969297
// MI455X (gfx1250) — compile-verified, matrix operand order checked by reading, not run on hardware
//
#include <hip/hip_runtime.h>
#include <hip/hip_bf16.h>
#include <math.h>

#define Bn 8
#define Nn 8192
#define Mn 1024
#define Kn 32
#define Cn 64
#define Rn (Bn * Mn * Kn)   // 262144 rows through the MLP
#define INFF __builtin_inff()

typedef _Float16 h8   __attribute__((ext_vector_type(8)));
typedef _Float16 v16h __attribute__((ext_vector_type(16)));
typedef float    v8f  __attribute__((ext_vector_type(8)));

__device__ __forceinline__ float gelu_exact(float x) {
    return 0.5f * x * (1.0f + erff(x * 0.70710678118654752440f));
}

// ---------------------------------------------------------------------------
// 1) Farthest point sampling: 1 block per batch, 1024 threads, 8 points/thread
// ---------------------------------------------------------------------------
__global__ void fps_kernel(const float* __restrict__ xyz,
                           float* __restrict__ newxyz_ws,
                           float* __restrict__ out_xyz) {
    int b = blockIdx.x;
    int t = threadIdx.x;
    const float* X = xyz + (size_t)b * Nn * 3;

    float px[8], py[8], pz[8], dm[8];
#pragma unroll
    for (int j = 0; j < 8; ++j) {
        int p = t * 8 + j;
        px[j] = X[p * 3 + 0];
        py[j] = X[p * 3 + 1];
        pz[j] = X[p * 3 + 2];
        dm[j] = INFF;
    }

    __shared__ float rv[1024];
    __shared__ int   ri[1024];
    __shared__ float cen[3];

    int far = 0;
    for (int m = 0; m < Mn; ++m) {
        if (t == 0) {
            float cx = X[far * 3 + 0], cy = X[far * 3 + 1], cz = X[far * 3 + 2];
            cen[0] = cx; cen[1] = cy; cen[2] = cz;
            size_t o = ((size_t)b * Mn + m) * 3;
            newxyz_ws[o + 0] = cx; newxyz_ws[o + 1] = cy; newxyz_ws[o + 2] = cz;
            out_xyz[o + 0] = cx; out_xyz[o + 1] = cy; out_xyz[o + 2] = cz;
        }
        __syncthreads();
        float cx = cen[0], cy = cen[1], cz = cen[2];

        float bv = -1.0f; int bi = 0;
#pragma unroll
        for (int j = 0; j < 8; ++j) {
            float dx = px[j] - cx, dy = py[j] - cy, dz = pz[j] - cz;
            float d = dx * dx + dy * dy + dz * dz;
            dm[j] = fminf(dm[j], d);
            if (dm[j] > bv) { bv = dm[j]; bi = t * 8 + j; }
        }
        rv[t] = bv; ri[t] = bi;
        __syncthreads();
        for (int s = 512; s > 0; s >>= 1) {
            if (t < s) {
                float v2 = rv[t + s]; int i2 = ri[t + s];
                if (v2 > rv[t] || (v2 == rv[t] && i2 < ri[t])) { rv[t] = v2; ri[t] = i2; }
            }
            __syncthreads();
        }
        far = ri[0];
        __syncthreads();
    }
}

// ---------------------------------------------------------------------------
// 2) Ball query: 1 block (256 thr) per centroid; 32 lexicographic min passes
// ---------------------------------------------------------------------------
__global__ void ball_query_kernel(const float* __restrict__ xyz,
                                  const float* __restrict__ newxyz_ws,
                                  int* __restrict__ gidx) {
    __shared__ float sd[Nn];
    __shared__ float rv[256];
    __shared__ int   ri[256];

    int bm = blockIdx.x;
    int b  = bm / Mn;
    int t  = threadIdx.x;
    const float* X = xyz + (size_t)b * Nn * 3;
    float cx = newxyz_ws[(size_t)bm * 3 + 0];
    float cy = newxyz_ws[(size_t)bm * 3 + 1];
    float cz = newxyz_ws[(size_t)bm * 3 + 2];

    for (int p = t; p < Nn; p += 256) {
        float dx = cx - X[p * 3 + 0];
        float dy = cy - X[p * 3 + 1];
        float dz = cz - X[p * 3 + 2];
        float d = sqrtf(dx * dx + dy * dy + dz * dz);
        sd[p] = (d > 0.5f) ? INFF : d;
    }
    __syncthreads();

    float lastv = -INFF; int lasti = -1; int idx0 = 0;
    for (int kk = 0; kk < Kn; ++kk) {
        float v = INFF; int i = Nn;   // sentinel loses ties on index
        for (int j = 0; j < 32; ++j) {
            int p = t * 32 + j;
            float val = sd[p];
            if (val > lastv || (val == lastv && p > lasti)) {
                if (val < v || (val == v && p < i)) { v = val; i = p; }
            }
        }
        rv[t] = v; ri[t] = i;
        __syncthreads();
        for (int s = 128; s > 0; s >>= 1) {
            if (t < s) {
                float v2 = rv[t + s]; int i2 = ri[t + s];
                if (v2 < rv[t] || (v2 == rv[t] && i2 < ri[t])) { rv[t] = v2; ri[t] = i2; }
            }
            __syncthreads();
        }
        float bv = rv[0]; int bi = ri[0];
        if (kk == 0) idx0 = bi;
        if (t == 0) gidx[(size_t)bm * Kn + kk] = (bv == INFF) ? idx0 : bi;
        lastv = bv; lasti = bi;
        __syncthreads();
    }
}

// ---------------------------------------------------------------------------
// 3) Gather + relative xyz + concat, padded f16 rows of 96 channels
// ---------------------------------------------------------------------------
__global__ void gather_kernel(const float* __restrict__ xyz,
                              const float* __restrict__ feat,
                              const float* __restrict__ newxyz_ws,
                              const int* __restrict__ gidx,
                              _Float16* __restrict__ h0) {
    int i = blockIdx.x * blockDim.x + threadIdx.x;      // over Rn*96
    int r = i / 96;
    int c = i % 96;
    int bm = r / Kn;
    int b = bm / Mn;
    int p = gidx[r];
    float v = 0.0f;
    if (c < 3) {
        v = xyz[((size_t)b * Nn + p) * 3 + c] - newxyz_ws[(size_t)bm * 3 + c];
    } else if (c < 67) {
        v = feat[((size_t)b * Nn + p) * Cn + (c - 3)];
    }
    h0[(size_t)r * 96 + c] = (_Float16)v;
}

// ---------------------------------------------------------------------------
// 4) Weight prep: f32 [Cout,Cin] -> f16 [Cout, CinPad] zero-padded
// ---------------------------------------------------------------------------
__global__ void wprep_kernel(const float* __restrict__ W, _Float16* __restrict__ Wh,
                             int Cout, int Cin, int CinPad) {
    int i = blockIdx.x * blockDim.x + threadIdx.x;
    if (i >= Cout * CinPad) return;
    int o = i / CinPad, c = i % CinPad;
    Wh[i] = (c < Cin) ? (_Float16)W[(size_t)o * Cin + c] : (_Float16)0.0f;
}

// ---------------------------------------------------------------------------
// 5) WMMA GEMM: one wave per 16-row tile, NT 16-col tiles, K-loop of 32 (f16)
//    A: [Rn, KT*32] f16   Wh: [NT*16, KT*32] f16   Hpre: [Rn, NT*16] f16
// ---------------------------------------------------------------------------
template <int KT, int NT>
__global__ void gemm_kernel(const _Float16* __restrict__ A,
                            const _Float16* __restrict__ Wh,
                            const float* __restrict__ bias,
                            _Float16* __restrict__ Hpre) {
    const int CINPAD = KT * 32;
    const int COUT   = NT * 16;
    int lane = threadIdx.x & 31;
    int wid  = blockIdx.x * (blockDim.x >> 5) + (threadIdx.x >> 5);
    int r0   = wid * 16;
    int hi   = lane >> 4;          // half-wave selects K sub-groups
    int ln   = lane & 15;

    const _Float16* Arow = A + (size_t)(r0 + ln) * CINPAD;

    v8f acc[NT] = {};
#pragma unroll
    for (int kt = 0; kt < KT; ++kt) {
        // A fragment: 16x32 f16; lanes 0-15 hold K {0..7,16..23}, 16-31 hold {8..15,24..31}
        h8 alo = *(const h8*)(Arow + kt * 32 + hi * 8);
        h8 ahi = *(const h8*)(Arow + kt * 32 + 16 + hi * 8);
        if (kt + 1 < KT) __builtin_prefetch(Arow + (kt + 1) * 32, 0, 1);
        v16h a;
#pragma unroll
        for (int e = 0; e < 8; ++e) { a[e] = alo[e]; a[e + 8] = ahi[e]; }
#pragma unroll
        for (int nt = 0; nt < NT; ++nt) {
            const _Float16* Wrow = Wh + (size_t)(nt * 16 + ln) * CINPAD;
            h8 blo = *(const h8*)(Wrow + kt * 32 + hi * 8);
            h8 bhi = *(const h8*)(Wrow + kt * 32 + 16 + hi * 8);
            v16h bfrag;
#pragma unroll
            for (int e = 0; e < 8; ++e) { bfrag[e] = blo[e]; bfrag[e + 8] = bhi[e]; }
            acc[nt] = __builtin_amdgcn_wmma_f32_16x16x32_f16(
                false, a, false, bfrag, (short)0, acc[nt], false, false);
        }
    }

    // Epilogue: +bias, store f16.  D layout: lane&15 = col, hi selects rows 0-7 / 8-15
#pragma unroll
    for (int nt = 0; nt < NT; ++nt) {
        int o = nt * 16 + ln;
        float bia = bias[o];
        int rbase = r0 + (hi ? 8 : 0);
#pragma unroll
        for (int rr = 0; rr < 8; ++rr) {
            float v = acc[nt][rr] + bia;
            Hpre[(size_t)(rbase + rr) * COUT + o] = (_Float16)v;
        }
    }
}

// ---------------------------------------------------------------------------
// 6) BN stats: deterministic two-stage (per-block partials, then finalize)
// ---------------------------------------------------------------------------
__global__ void stats_kernel(const _Float16* __restrict__ H,
                             float* __restrict__ partials, int Cout) {
    __shared__ float ss[256];
    __shared__ float qq[256];
    int t = threadIdx.x;
    int c = t % Cout;
    int g = t / Cout;
    int RG = 256 / Cout;
    float s = 0.0f, q = 0.0f;
    for (size_t r = (size_t)blockIdx.x * RG + g; r < Rn; r += (size_t)gridDim.x * RG) {
        float x = (float)H[r * Cout + c];
        s += x; q += x * x;
    }
    ss[t] = s; qq[t] = q;
    __syncthreads();
    if (t < Cout) {
        for (int gg = 1; gg < RG; ++gg) { s += ss[t + gg * Cout]; q += qq[t + gg * Cout]; }
        partials[((size_t)blockIdx.x * 128 + t) * 2 + 0] = s;
        partials[((size_t)blockIdx.x * 128 + t) * 2 + 1] = q;
    }
}

__global__ void finalize_kernel(const float* __restrict__ partials,
                                float* __restrict__ stats, int Cout) {
    int c = threadIdx.x;
    if (c >= Cout) return;
    float s = 0.0f, q = 0.0f;
    for (int i = 0; i < 1024; ++i) {
        s += partials[((size_t)i * 128 + c) * 2 + 0];
        q += partials[((size_t)i * 128 + c) * 2 + 1];
    }
    const float invCnt = 1.0f / (float)Rn;
    float mu = s * invCnt;
    float var = q * invCnt - mu * mu;
    stats[c]       = mu;
    stats[128 + c] = rsqrtf(var + 1e-5f);
}

// ---------------------------------------------------------------------------
// 7) BN + exact GELU (intermediate layers)
// ---------------------------------------------------------------------------
__global__ void act_kernel(const _Float16* __restrict__ Hpre,
                           _Float16* __restrict__ Hact,
                           const float* __restrict__ stats,
                           const float* __restrict__ gamma,
                           const float* __restrict__ beta, int Cout) {
    size_t i = (size_t)blockIdx.x * blockDim.x + threadIdx.x;
    int c = (int)(i % Cout);
    float x = (float)Hpre[i];
    float y = (x - stats[c]) * stats[128 + c] * gamma[c] + beta[c];
    Hact[i] = (_Float16)gelu_exact(y);
}

// ---------------------------------------------------------------------------
// 8) Final: BN + GELU + max over K -> out features [B*M, 128]
// ---------------------------------------------------------------------------
__global__ void final_kernel(const _Float16* __restrict__ H3,
                             const float* __restrict__ stats,
                             const float* __restrict__ gamma,
                             const float* __restrict__ beta,
                             float* __restrict__ outfeat) {
    int bm = blockIdx.x;
    int c  = threadIdx.x;
    float mu = stats[c], rs = stats[128 + c], ga = gamma[c], be = beta[c];
    float mv = -INFF;
    for (int k = 0; k < Kn; ++k) {
        float x = (float)H3[((size_t)bm * Kn + k) * 128 + c];
        float y = gelu_exact((x - mu) * rs * ga + be);
        mv = fmaxf(mv, y);
    }
    outfeat[(size_t)bm * 128 + c] = mv;
}

// ---------------------------------------------------------------------------
// Host: sequence everything on `stream`, bump-allocate workspace
// ---------------------------------------------------------------------------
extern "C" void kernel_launch(void* const* d_in, const int* in_sizes, int n_in,
                              void* d_out, int out_size, void* d_ws, size_t ws_size,
                              hipStream_t stream) {
    const float* xyz  = (const float*)d_in[0];
    const float* feat = (const float*)d_in[1];
    const float* W1 = (const float*)d_in[2];
    const float* b1 = (const float*)d_in[3];
    const float* g1 = (const float*)d_in[4];
    const float* bt1 = (const float*)d_in[5];
    const float* W2 = (const float*)d_in[6];
    const float* b2 = (const float*)d_in[7];
    const float* g2 = (const float*)d_in[8];
    const float* bt2 = (const float*)d_in[9];
    const float* W3 = (const float*)d_in[10];
    const float* b3 = (const float*)d_in[11];
    const float* g3 = (const float*)d_in[12];
    const float* bt3 = (const float*)d_in[13];
    float* out = (float*)d_out;

    char* ws = (char*)d_ws;
    size_t off = 0;
    auto alc = [&](size_t n) { size_t o = off; off = (off + n + 255) & ~(size_t)255; return o; };

    float*    newxyz   = (float*)(ws + alc((size_t)Bn * Mn * 3 * sizeof(float)));
    int*      gidx     = (int*)(ws + alc((size_t)Rn * sizeof(int)));
    _Float16* h0       = (_Float16*)(ws + alc((size_t)Rn * 96 * sizeof(_Float16)));
    _Float16* wh1      = (_Float16*)(ws + alc((size_t)64 * 96 * sizeof(_Float16)));
    _Float16* wh2      = (_Float16*)(ws + alc((size_t)64 * 64 * sizeof(_Float16)));
    _Float16* wh3      = (_Float16*)(ws + alc((size_t)128 * 64 * sizeof(_Float16)));
    _Float16* hpre     = (_Float16*)(ws + alc((size_t)Rn * 128 * sizeof(_Float16)));
    _Float16* hact     = (_Float16*)(ws + alc((size_t)Rn * 64 * sizeof(_Float16)));
    float*    partials = (float*)(ws + alc((size_t)1024 * 128 * 2 * sizeof(float)));
    float*    st1      = (float*)(ws + alc(256 * sizeof(float)));
    float*    st2      = (float*)(ws + alc(256 * sizeof(float)));
    float*    st3      = (float*)(ws + alc(256 * sizeof(float)));

    // 1) FPS: centroids + new_xyz (also written straight into d_out)
    fps_kernel<<<Bn, 1024, 0, stream>>>(xyz, newxyz, out);

    // 2) Ball query (K nearest within radius, nearest-fallback)
    ball_query_kernel<<<Bn * Mn, 256, 0, stream>>>(xyz, newxyz, gidx);

    // 3) Gather + concat into padded f16 rows
    gather_kernel<<<(Rn * 96) / 256, 256, 0, stream>>>(xyz, feat, newxyz, gidx, h0);

    // 4) Weight prep
    wprep_kernel<<<(64 * 96 + 255) / 256, 256, 0, stream>>>(W1, wh1, 64, 67, 96);
    wprep_kernel<<<(64 * 64 + 255) / 256, 256, 0, stream>>>(W2, wh2, 64, 64, 64);
    wprep_kernel<<<(128 * 64 + 255) / 256, 256, 0, stream>>>(W3, wh3, 128, 64, 64);

    // Layer 1: 96 -> 64
    gemm_kernel<3, 4><<<Rn / 128, 256, 0, stream>>>(h0, wh1, b1, hpre);
    stats_kernel<<<1024, 256, 0, stream>>>(hpre, partials, 64);
    finalize_kernel<<<1, 128, 0, stream>>>(partials, st1, 64);
    act_kernel<<<(Rn * 64) / 256, 256, 0, stream>>>(hpre, hact, st1, g1, bt1, 64);

    // Layer 2: 64 -> 64
    gemm_kernel<2, 4><<<Rn / 128, 256, 0, stream>>>(hact, wh2, b2, hpre);
    stats_kernel<<<1024, 256, 0, stream>>>(hpre, partials, 64);
    finalize_kernel<<<1, 128, 0, stream>>>(partials, st2, 64);
    act_kernel<<<(Rn * 64) / 256, 256, 0, stream>>>(hpre, hact, st2, g2, bt2, 64);

    // Layer 3: 64 -> 128, then BN+GELU+maxpool fused into output
    gemm_kernel<2, 8><<<Rn / 128, 256, 0, stream>>>(hact, wh3, b3, hpre);
    stats_kernel<<<1024, 256, 0, stream>>>(hpre, partials, 128);
    finalize_kernel<<<1, 128, 0, stream>>>(partials, st3, 128);
    final_kernel<<<Bn * Mn, 128, 0, stream>>>(hpre, st3, g3, bt3, out + (size_t)Bn * Mn * 3);
}
